// ScaledDotProdAttn_19928648253501
// MI455X (gfx1250) — matrix operand order repaired and checked by reading, not compiled, not run on hardware
//
#include <hip/hip_runtime.h>

typedef __attribute__((ext_vector_type(16))) _Float16 v16h;
typedef __attribute__((ext_vector_type(8)))  float    v8f;
typedef unsigned int u32x4 __attribute__((ext_vector_type(4)));
typedef int          i32x8 __attribute__((ext_vector_type(8)));
typedef int          i32x4 __attribute__((ext_vector_type(4)));
typedef _Float16 half_t;

#define LQ 2048
#define LK 2048
#define DH 64      // head dim
#define BQ 128     // q rows per block
#define BK 64      // k rows per iteration
#define NWAVE 8

// ---- dynamic-LDS layout (bytes) -------------------------------------------
// raw f32 staging is double buffered and filled by the Tensor Data Mover.
#define RAW_K(buf)  ((buf) * 32768u)            // 64x64 f32 = 16384 B
#define RAW_V(buf)  ((buf) * 32768u + 16384u)   // 64x64 f32 = 16384 B
#define SK_OFF      65536u                      // 64x64 f16 = 8192 B
#define SVT_OFF     73728u                      // 64x64 f16 = 8192 B
#define SP_OFF      81920u                      // 8x16x64 f16 = 16384 B
#define SMEM_BYTES  98304u

// ---------------------------------------------------------------------------
// Flash attention forward, causal, fp32 in/out, f16 WMMA compute.
// One block = 8 waves = 128 q rows of one batch. Wave w owns rows
// [qbase + 16w, qbase + 16w + 16). K/V tiles of 64 rows are DMA'd into LDS by
// the TDM (double buffered, TENSORcnt-tracked), converted to f16 in LDS, and
// consumed by v_wmma_f32_16x16x32_f16 with an online softmax.
// ---------------------------------------------------------------------------
__global__ __launch_bounds__(256, 2)
void ScaledDotProdAttn_fa_kernel(const float* __restrict__ Q,
                                 const float* __restrict__ K,
                                 const float* __restrict__ V,
                                 float* __restrict__ O)
{
    extern __shared__ char smem[];
    half_t* sK  = (half_t*)(smem + SK_OFF);    // [64][64] K tile, row-major
    half_t* sVt = (half_t*)(smem + SVT_OFF);   // [64][64] V tile, transposed
    half_t* sP  = (half_t*)(smem + SP_OFF);    // [8][16][64] per-wave P

    const int tid   = threadIdx.x;
    const int wave  = tid >> 5;
    const int lane  = tid & 31;
    const int lhalf = lane >> 4;             // 0: lanes 0-15, 1: lanes 16-31
    const int l16   = lane & 15;

    const int b     = blockIdx.x >> 4;       // LQ/BQ == 16 tiles per batch
    const int qt    = blockIdx.x & 15;
    const int qbase = qt * BQ;
    const int q0w   = qbase + wave * 16;     // first q row of this wave

    const float scale = 0.03125f;            // 1/sqrt(D_MODEL=1024)

    const float* Qb = Q + (size_t)b * LQ * DH;
    const float* Kb = K + (size_t)b * LK * DH;
    const float* Vb = V + (size_t)b * LK * DH;
    float*       Ob = O + (size_t)b * LQ * DH;

    // ---- TDM descriptor issue: 64x64 f32 tile of a [2048][64] tensor ------
    auto tdm_tile_to_lds = [&](const float* tile_ptr, unsigned lds_off) {
        const unsigned long long ga = (unsigned long long)(uintptr_t)tile_ptr;
        u32x4 g0;
        g0[0] = 1u;                                   // count=1 (valid), user D#
        g0[1] = lds_off;                              // lds_addr (bytes)
        g0[2] = (unsigned)(ga & 0xFFFFFFFFu);         // global_addr[31:0]
        g0[3] = (unsigned)((ga >> 32) & 0x1FFFFFFu)   // global_addr[56:32]
              | (2u << 30);                           // type = 2 ("image")
        i32x8 g1;
        g1[0] = (int)(2u << 16);                      // data_size = 2 -> 4 bytes
        g1[1] = (int)(64u << 16);                     // tensor_dim0 = 64 (bits 79:48)
        g1[2] = (int)(2048u << 16);                   // tensor_dim1 = 2048 (bits 111:80)
        g1[3] = (int)(64u << 16);                     // tile_dim0 = 64 (bits 127:112)
        g1[4] = 64;                                   // tile_dim1 = 64, tile_dim2 = 0
        g1[5] = 64;                                   // tensor_dim0_stride = 64
        g1[6] = 0;                                    // dim0_stride[47:32], dim1_stride lo
        g1[7] = 0;
        i32x4 g2 = {0, 0, 0, 0};                      // 2D tensor: unused
        i32x4 g3 = {0, 0, 0, 0};
        i32x8 g4 = {0, 0, 0, 0, 0, 0, 0, 0};          // 6-arg toolchain extra group
        __builtin_amdgcn_tensor_load_to_lds(g0, g1, g2, g3, g4, 0);
    };

    // ---- Q strip -> A-operand registers (scaled, f16) ---------------------
    // A 16-bit 16x32 layout: lane holds row l16; lane-half h owns K-halves
    // {8h..8h+7} and {16+8h..16+8h+7}.  aq0 covers d=0..31, aq1 d=32..63.
    v16h aq0{}, aq1{};
    {
        const float* qrow = Qb + (size_t)(q0w + l16) * DH + 8 * lhalf;
        #pragma unroll
        for (int i = 0; i < 8; ++i) {
            aq0[i]     = (half_t)(qrow[i]      * scale);
            aq0[8 + i] = (half_t)(qrow[16 + i] * scale);
            aq1[i]     = (half_t)(qrow[32 + i] * scale);
            aq1[8 + i] = (half_t)(qrow[48 + i] * scale);
        }
    }

    // ---- running state ----------------------------------------------------
    v8f   oacc[4];                           // O strip: 4 d-tiles of 16x16 f32
    float mrow[8], lrow[8];
    #pragma unroll
    for (int t = 0; t < 4; ++t) oacc[t] = v8f{};
    #pragma unroll
    for (int r = 0; r < 8; ++r) { mrow[r] = -3.0e38f; lrow[r] = 0.0f; }

    const int ktiles = (qbase + BQ) / BK;    // causal upper bound for block

    // ---- prologue: DMA tile 0 into raw buffer 0 ---------------------------
    if (wave == 0) {
        tdm_tile_to_lds(Kb, RAW_K(0));
        tdm_tile_to_lds(Vb, RAW_V(0));
    }

    for (int kt = 0; kt < ktiles; ++kt) {
        const int cur   = kt & 1;
        const int kbase = kt * BK;

        if (wave == 0) __builtin_amdgcn_s_wait_tensorcnt(0);  // raw[cur] done
        __syncthreads();                                      // publish to WG

        // DMA next tile into the other raw buffer while we compute this one.
        if (wave == 0 && (kt + 1) < ktiles) {
            const int nb = cur ^ 1;
            tdm_tile_to_lds(Kb + (size_t)(kbase + BK) * DH, RAW_K(nb));
            tdm_tile_to_lds(Vb + (size_t)(kbase + BK) * DH, RAW_V(nb));
        }

        // ---- convert raw f32 tile -> f16 operand buffers (K rowmajor, V^T)
        {
            const int row = tid >> 2;                 // 0..63
            const int c0  = (tid & 3) * 16;           // 0,16,32,48
            const float* kr = (const float*)(smem + RAW_K(cur)) + row * DH + c0;
            const float* vr = (const float*)(smem + RAW_V(cur)) + row * DH + c0;
            #pragma unroll
            for (int i = 0; i < 16; i += 4) {
                float4 k4 = *(const float4*)(kr + i);
                float4 v4 = *(const float4*)(vr + i);
                sK[row * DH + c0 + i + 0] = (half_t)k4.x;
                sK[row * DH + c0 + i + 1] = (half_t)k4.y;
                sK[row * DH + c0 + i + 2] = (half_t)k4.z;
                sK[row * DH + c0 + i + 3] = (half_t)k4.w;
                sVt[(c0 + i + 0) * BK + row] = (half_t)v4.x;
                sVt[(c0 + i + 1) * BK + row] = (half_t)v4.y;
                sVt[(c0 + i + 2) * BK + row] = (half_t)v4.z;
                sVt[(c0 + i + 3) * BK + row] = (half_t)v4.w;
            }
        }
        __syncthreads();                              // sK/sVt ready

        // Tile fully above the diagonal for this wave -> skip compute.
        if (kbase <= q0w + 15) {
            // ---- S = Q K^T : 4 column tiles, K-dim 64 via 2 chained WMMAs
            v8f sacc[4];
            #pragma unroll
            for (int t = 0; t < 4; ++t) {
                // B 16-bit 32x16 layout: lane-half h holds K-rows {8h..8h+7}
                // and {16+8h..16+8h+7} of column N=l16, matching A.
                v16h bk0{}, bk1{};
                const half_t* krow = sK + (t * 16 + l16) * DH + 8 * lhalf;
                #pragma unroll
                for (int i = 0; i < 16; ++i) {
                    bk0[i] = krow[i + (i & 8)];        // d = 8h + i (+8 for i>=8)      (d 0..31)
                    bk1[i] = krow[32 + i + (i & 8)];   // d = 32 + 8h + i (+8 for i>=8) (d 32..63)
                }
                v8f c = {};
                c = __builtin_amdgcn_wmma_f32_16x16x32_f16(false, aq0, false, bk0,
                                                           (short)0, c, false, false);
                c = __builtin_amdgcn_wmma_f32_16x16x32_f16(false, aq1, false, bk1,
                                                           (short)0, c, false, false);
                sacc[t] = c;
            }

            // ---- causal mask (element granularity, only on diagonal tiles)
            if (kbase + BK - 1 > q0w) {
                #pragma unroll
                for (int t = 0; t < 4; ++t) {
                    #pragma unroll
                    for (int r = 0; r < 8; ++r) {
                        const int qrw = q0w + r + 8 * lhalf;     // C row M
                        const int kcl = kbase + t * 16 + l16;    // C col N
                        if (kcl > qrw) sacc[t][r] = -1.0e9f;
                    }
                }
            }

            // ---- online softmax: row reductions across 16 lanes per half
            float corr[8];
            #pragma unroll
            for (int r = 0; r < 8; ++r) {
                float mx = fmaxf(fmaxf(sacc[0][r], sacc[1][r]),
                                 fmaxf(sacc[2][r], sacc[3][r]));
                mx = fmaxf(mx, __shfl_xor(mx, 1, 32));
                mx = fmaxf(mx, __shfl_xor(mx, 2, 32));
                mx = fmaxf(mx, __shfl_xor(mx, 4, 32));
                mx = fmaxf(mx, __shfl_xor(mx, 8, 32));
                const float mnew = fmaxf(mrow[r], mx);
                corr[r] = __expf(mrow[r] - mnew);
                mrow[r] = mnew;
                float rs = 0.0f;
                #pragma unroll
                for (int t = 0; t < 4; ++t) {
                    const float p = __expf(sacc[t][r] - mnew);
                    sacc[t][r] = p;
                    rs += p;
                }
                rs += __shfl_xor(rs, 1, 32);
                rs += __shfl_xor(rs, 2, 32);
                rs += __shfl_xor(rs, 4, 32);
                rs += __shfl_xor(rs, 8, 32);
                lrow[r] = lrow[r] * corr[r] + rs;
            }

            // ---- rescale O; park P (f16) in per-wave LDS for re-layout ---
            #pragma unroll
            for (int t = 0; t < 4; ++t) {
                #pragma unroll
                for (int r = 0; r < 8; ++r) {
                    oacc[t][r] *= corr[r];
                    sP[(wave * 16 + r + 8 * lhalf) * BK + t * 16 + l16] =
                        (half_t)sacc[t][r];
                }
            }
            // DS ops from one wave are in-order: stores above are visible to
            // the loads below without a workgroup barrier (per-wave buffer).

            // ---- A operand for P (K-dim = k-rows 0..63) ------------------
            v16h ap0{}, ap1{};
            {
                const half_t* prow = sP + (wave * 16 + l16) * BK + 8 * lhalf;
                #pragma unroll
                for (int i = 0; i < 8; ++i) {
                    ap0[i]     = prow[i];
                    ap0[8 + i] = prow[16 + i];
                    ap1[i]     = prow[32 + i];
                    ap1[8 + i] = prow[48 + i];
                }
            }

            // ---- O += P V : 4 d-tiles, K-dim 64 via 2 chained WMMAs ------
            #pragma unroll
            for (int t = 0; t < 4; ++t) {
                v16h bv0{}, bv1{};
                const half_t* vcol = sVt + (t * 16 + l16) * BK + 8 * lhalf;
                #pragma unroll
                for (int i = 0; i < 16; ++i) {
                    bv0[i] = vcol[i + (i & 8)];        // k-rows 8h+i (+8 for i>=8)      (n 0..31)
                    bv1[i] = vcol[32 + i + (i & 8)];   // k-rows 32+8h+i (+8 for i>=8)   (n 32..63)
                }
                oacc[t] = __builtin_amdgcn_wmma_f32_16x16x32_f16(
                    false, ap0, false, bv0, (short)0, oacc[t], false, false);
                oacc[t] = __builtin_amdgcn_wmma_f32_16x16x32_f16(
                    false, ap1, false, bv1, (short)0, oacc[t], false, false);
            }
        }
        __syncthreads();   // protect sK/sVt (and raw[cur]) before next stage
    }

    // ---- epilogue: O / l, fp32 out ----------------------------------------
    #pragma unroll
    for (int t = 0; t < 4; ++t) {
        #pragma unroll
        for (int r = 0; r < 8; ++r) {
            const int qr = q0w + r + 8 * lhalf;
            Ob[(size_t)qr * DH + t * 16 + l16] = oacc[t][r] / lrow[r];
        }
    }
}

extern "C" void kernel_launch(void* const* d_in, const int* in_sizes, int n_in,
                              void* d_out, int out_size, void* d_ws, size_t ws_size,
                              hipStream_t stream) {
    const float* q = (const float*)d_in[0];
    const float* k = (const float*)d_in[1];
    const float* v = (const float*)d_in[2];
    // d_in[3] is the boolean causal mask; it is purely causal and recomputed
    // analytically inside the kernel, so it is never read.
    float* out = (float*)d_out;

    const int B = 16;
    const dim3 grid(B * (LQ / BQ));   // 256 workgroups
    const dim3 block(256);            // 8 waves (wave32)
    ScaledDotProdAttn_fa_kernel<<<grid, block, SMEM_BYTES, stream>>>(q, k, v, out);
}
